// GridCLIPModel_3504693314033
// MI455X (gfx1250) — compile-verified, matrix operand order checked by reading, not run on hardware
//
#include <hip/hip_runtime.h>
#include <hip/hip_bf16.h>

typedef __attribute__((ext_vector_type(16))) _Float16 v16h;
typedef __attribute__((ext_vector_type(8)))  _Float16 v8h;
typedef __attribute__((ext_vector_type(8)))  float    v8f;
typedef __attribute__((ext_vector_type(4)))  unsigned int v4u;

// ---- level metadata (matches torch-ngp offset computation) ----
// l<3 dense (res+1 = 17,33,65), l>=3 hashed with hsize=2^19 (mask 0x7FFFF)
__constant__ unsigned c_off[16] = {
    0u, 4920u, 40864u, 315496u, 839784u, 1364072u, 1888360u, 2412648u,
    2936936u, 3461224u, 3985512u, 4509800u, 5034088u, 5558376u, 6082664u, 6606952u
};

// ---------------- f32 -> f16 weight convert ----------------
__global__ void f32_to_f16_kernel(const float* __restrict__ src,
                                  _Float16* __restrict__ dst, int n) {
    int i = blockIdx.x * blockDim.x + threadIdx.x;
    if (i < n) dst[i] = (_Float16)src[i];
}

// ---------------- grid encode + affordance embedding ----------------
__global__ __launch_bounds__(256)
void encode_kernel(const float* __restrict__ x,
                   const float* __restrict__ aff,
                   const float* __restrict__ table,
                   const float* __restrict__ Wa,
                   const float* __restrict__ ba,
                   _Float16* __restrict__ comb, int N) {
    int n = blockIdx.x * blockDim.x + threadIdx.x;
    if (n >= N) return;

    float px = x[n * 3 + 0], py = x[n * 3 + 1], pz = x[n * 3 + 2];
    // bounded = (x+10)/20 ; x01 = (bounded+1)/2
    float x0 = ((px + 10.0f) * 0.05f + 1.0f) * 0.5f;
    float y0 = ((py + 10.0f) * 0.05f + 1.0f) * 0.5f;
    float z0 = ((pz + 10.0f) * 0.05f + 1.0f) * 0.5f;
    float a = aff[n];

    _Float16* outrow = comb + (size_t)n * 128;

    for (int l = 0; l < 16; ++l) {
        float scale = (float)(16u << l) - 1.0f;
        bool dense = (l < 3);
        unsigned res1 = (16u << l) + 1u;
        unsigned res1sq = res1 * res1;
        unsigned off = c_off[l];

        float fx = x0 * scale + 0.5f, fy = y0 * scale + 0.5f, fz = z0 * scale + 0.5f;
        float gx = floorf(fx), gy = floorf(fy), gz = floorf(fz);
        float rx = fx - gx, ry = fy - gy, rz = fz - gz;
        unsigned bx = (unsigned)gx, by = (unsigned)gy, bz = (unsigned)gz;

        float acc[8];
        #pragma unroll
        for (int j = 0; j < 8; ++j) acc[j] = 0.0f;

        #pragma unroll
        for (int c = 0; c < 8; ++c) {
            unsigned ux = bx + (c & 1);
            unsigned uy = by + ((c >> 1) & 1);
            unsigned uz = bz + ((c >> 2) & 1);
            float w = ((c & 1)        ? rx : 1.0f - rx) *
                      (((c >> 1) & 1) ? ry : 1.0f - ry) *
                      (((c >> 2) & 1) ? rz : 1.0f - rz);
            unsigned idx;
            if (dense) {
                idx = ux + uy * res1 + uz * res1sq;   // < hsize, mod is identity
            } else {
                idx = (ux * 1u) ^ (uy * 2654435761u) ^ (uz * 805459861u);
                idx &= 0x7FFFFu;                       // hsize = 2^19
            }
            const float4* t = (const float4*)(table + (size_t)(idx + off) * 8u);
            float4 t0 = t[0], t1 = t[1];
            acc[0] = fmaf(w, t0.x, acc[0]); acc[1] = fmaf(w, t0.y, acc[1]);
            acc[2] = fmaf(w, t0.z, acc[2]); acc[3] = fmaf(w, t0.w, acc[3]);
            acc[4] = fmaf(w, t1.x, acc[4]); acc[5] = fmaf(w, t1.y, acc[5]);
            acc[6] = fmaf(w, t1.z, acc[6]); acc[7] = fmaf(w, t1.w, acc[7]);
        }

        v8h r;
        #pragma unroll
        for (int j = 0; j < 8; ++j) {
            int f = l * 8 + j;
            r[j] = (_Float16)(acc[j] + a * Wa[f] + ba[f]);
        }
        *(v8h*)(outrow + l * 8) = r;   // 16B store
    }
}

// ---------------- WMMA GEMM: out = act(A[N,IN] @ W[OUT,IN]^T + b) ----------------
// block: 256 threads (8 waves); 64 rows x 128 cols per block.
// wave w owns col-strip [w*16, w*16+16) and all four 16-row subtiles.
template<int IN, bool RELU, bool FINAL>
__global__ __launch_bounds__(256)
void gemm_kernel(const _Float16* __restrict__ A,
                 const _Float16* __restrict__ W,
                 const float* __restrict__ bias,
                 _Float16* __restrict__ outH,
                 float* __restrict__ outF,
                 int N, int OUTtotal) {
    extern __shared__ char smem[];
    _Float16* sA = (_Float16*)smem;                      // 64  x IN
    _Float16* sW = (_Float16*)(smem + 64 * IN * 2);      // 128 x IN

    const int tid = threadIdx.x;
    const int mbase = blockIdx.x * 64;
    const int nbase0 = blockIdx.y * 128;

    {   // stage A tile (coalesced 16B chunks)
        const v4u* g = (const v4u*)(A + (size_t)mbase * IN);
        v4u* s = (v4u*)sA;
        int rows = N - mbase; if (rows > 64) rows = 64;
        int nchunks = rows * (IN / 8);
        for (int i = tid; i < nchunks; i += 256) s[i] = g[i];
    }
    {   // stage W tile (128 rows of W, row-major [OUT][IN])
        const v4u* g = (const v4u*)(W + (size_t)nbase0 * IN);
        v4u* s = (v4u*)sW;
        for (int i = tid; i < 16 * IN; i += 256) s[i] = g[i];
    }
    __syncthreads();

    const int wave  = tid >> 5;
    const int lane  = tid & 31;
    const int khalf = lane >> 4;      // K-half select per ISA 16-bit A/B layout
    const int lrow  = lane & 15;
    const int ncol  = wave * 16 + lrow;       // within 128-col block tile

    float bn = bias[nbase0 + ncol];

    v8f zero8 = {0.f, 0.f, 0.f, 0.f, 0.f, 0.f, 0.f, 0.f};
    v8f acc[4];
    #pragma unroll
    for (int m = 0; m < 4; ++m) acc[m] = zero8;

    #pragma unroll
    for (int kk = 0; kk < IN; kk += 32) {
        // B fragment = column (ncol) of W^T = row ncol of W
        union { v16h v; v4u u[2]; } fb;
        const _Float16* pB = sW + (size_t)ncol * IN + kk + khalf * 8;
        fb.u[0] = *(const v4u*)pB;
        fb.u[1] = *(const v4u*)(pB + 16);
        #pragma unroll
        for (int m = 0; m < 4; ++m) {
            union { v16h v; v4u u[2]; } fa;
            const _Float16* pA = sA + (size_t)(m * 16 + lrow) * IN + kk + khalf * 8;
            fa.u[0] = *(const v4u*)pA;
            fa.u[1] = *(const v4u*)(pA + 16);
            acc[m] = __builtin_amdgcn_wmma_f32_16x16x32_f16(
                false, fa.v, false, fb.v, (short)0, acc[m], false, false);
        }
    }

    // C/D layout: acc[m][r] is element (row = m*16 + r + khalf*8, col = ncol)
    #pragma unroll
    for (int m = 0; m < 4; ++m) {
        #pragma unroll
        for (int r = 0; r < 8; ++r) {
            int row = mbase + m * 16 + r + khalf * 8;
            if (row >= N) continue;
            float val = acc[m][r] + bn;
            if (RELU) val = val > 0.0f ? val : 0.0f;
            if (!FINAL) {
                outH[(size_t)row * OUTtotal + nbase0 + ncol] = (_Float16)val;
            } else {
                // tuple split: image = cols [0,512), affordance = cols [512,1024)
                int nc = nbase0 + ncol;
                size_t off = (nc < 512)
                    ? ((size_t)row * 512 + nc)
                    : ((size_t)N * 512 + (size_t)row * 512 + (nc - 512));
                outF[off] = val;
            }
        }
    }
}

extern "C" void kernel_launch(void* const* d_in, const int* in_sizes, int n_in,
                              void* d_out, int out_size, void* d_ws, size_t ws_size,
                              hipStream_t stream) {
    const float* x     = (const float*)d_in[0];
    const float* aff   = (const float*)d_in[1];
    const float* table = (const float*)d_in[2];
    const float* Wa    = (const float*)d_in[3];
    const float* ba    = (const float*)d_in[4];
    // Wq/bq/Wk/bk (5..8) unused: seq_len==1 attention reduces to Wo(Wv x + bv)+bo
    const float* Wv = (const float*)d_in[9];
    const float* bv = (const float*)d_in[10];
    const float* Wo = (const float*)d_in[11];
    const float* bo = (const float*)d_in[12];
    const float* W1 = (const float*)d_in[13];
    const float* b1 = (const float*)d_in[14];
    const float* W2 = (const float*)d_in[15];
    const float* b2 = (const float*)d_in[16];
    const float* W3 = (const float*)d_in[17];
    const float* b3 = (const float*)d_in[18];

    const int N = in_sizes[1];           // 200000

    // workspace layout (bytes); h2 aliases comb+attn, h1 aliases v (all dead by then)
    char* ws = (char*)d_ws;
    _Float16* comb = (_Float16*)(ws);                         // [0, N*256)
    _Float16* attn = (_Float16*)(ws + (size_t)N * 256);       // [N*256, N*512)
    _Float16* vbuf = (_Float16*)(ws + (size_t)N * 512);       // [N*512, N*768)
    _Float16* h1   = (_Float16*)(ws + (size_t)N * 512);       // [N*512, N*1024)
    _Float16* h2   = (_Float16*)(ws);                         // [0, N*512)
    _Float16* Wv16 = (_Float16*)(ws + (size_t)N * 1024);
    _Float16* Wo16 = Wv16 + 128 * 128;
    _Float16* W116 = Wo16 + 128 * 128;
    _Float16* W216 = W116 + 256 * 128;
    _Float16* W316 = W216 + 256 * 256;

    // 1) weights -> f16
    f32_to_f16_kernel<<<(128 * 128 + 255) / 256, 256, 0, stream>>>(Wv, Wv16, 128 * 128);
    f32_to_f16_kernel<<<(128 * 128 + 255) / 256, 256, 0, stream>>>(Wo, Wo16, 128 * 128);
    f32_to_f16_kernel<<<(256 * 128 + 255) / 256, 256, 0, stream>>>(W1, W116, 256 * 128);
    f32_to_f16_kernel<<<(256 * 256 + 255) / 256, 256, 0, stream>>>(W2, W216, 256 * 256);
    f32_to_f16_kernel<<<(1024 * 256 + 255) / 256, 256, 0, stream>>>(W3, W316, 1024 * 256);

    // 2) grid encode + affordance embedding -> comb f16 [N,128]
    encode_kernel<<<(N + 255) / 256, 256, 0, stream>>>(x, aff, table, Wa, ba, comb, N);

    const int mblocks = (N + 63) / 64;       // 3125 (exact)
    const size_t sm128 = (size_t)(64 + 128) * 128 * 2;   // 48 KB
    const size_t sm256 = (size_t)(64 + 128) * 256 * 2;   // 96 KB

    // 3) v    = comb @ Wv^T + bv
    gemm_kernel<128, false, false><<<dim3(mblocks, 1), 256, sm128, stream>>>(
        comb, Wv16, bv, vbuf, nullptr, N, 128);
    // 4) attn = v @ Wo^T + bo
    gemm_kernel<128, false, false><<<dim3(mblocks, 1), 256, sm128, stream>>>(
        vbuf, Wo16, bo, attn, nullptr, N, 128);
    // 5) h1 = relu(attn @ W1^T + b1)
    gemm_kernel<128, true, false><<<dim3(mblocks, 2), 256, sm128, stream>>>(
        attn, W116, b1, h1, nullptr, N, 256);
    // 6) h2 = relu(h1 @ W2^T + b2)
    gemm_kernel<256, true, false><<<dim3(mblocks, 2), 256, sm256, stream>>>(
        h1, W216, b2, h2, nullptr, N, 256);
    // 7) out = h2 @ W3^T + b3  -> f32, split into (image, affordance) halves
    gemm_kernel<256, false, true><<<dim3(mblocks, 8), 256, sm256, stream>>>(
        h2, W316, b3, nullptr, (float*)d_out, N, 1024);
}
